// SelfAttention_22591527977643
// MI455X (gfx1250) — compile-verified, matrix operand order checked by reading, not run on hardware
//
#include <hip/hip_runtime.h>

// ---------------------------------------------------------------------------
// CDNA5 (gfx1250) self-attention forward, f16 WMMA with f32 accumulation.
// Fixed problem: B=2, T=2048, D=1024, H=16, hd=64 (per reference setup).
// Attention computed in transposed form (S^T = K·Q^T, O^T = V^T·P^T) so that
// softmax row statistics are per-lane and P converts C-frag -> B-frag in
// registers (no LDS round trip, no bpermute reduction chains).
// ---------------------------------------------------------------------------

typedef __attribute__((ext_vector_type(16))) _Float16 v16h;
typedef __attribute__((ext_vector_type(8)))  _Float16 v8h;
typedef __attribute__((ext_vector_type(8)))  float    v8f;
typedef __attribute__((ext_vector_type(4)))  float    v4f;

union Frag16 {
    v16h v;
    struct { v8h lo; v8h hi; } p;
};

// D = A x B + C  (16x16x32, f16 in, f32 acc)
__device__ __forceinline__ v8f wmma_f16(v16h a, v16h b, v8f c) {
    return __builtin_amdgcn_wmma_f32_16x16x32_f16(
        /*neg_a=*/false, a, /*neg_b=*/false, b,
        /*c_mod=*/(short)0, c, /*reuse_a=*/false, /*reuse_b=*/false);
}

// Load a 16x32 A-style fragment from a row-major matrix (also the layout of
// the B operand of A·B^T when the second matrix is stored row-major N x K).
// CDNA5 16-bit A layout:
//   lanes 0-15 : row = lane,    halves 0-7 -> K=k0+0..7,  halves 8-15 -> K=k0+16..23
//   lanes 16-31: row = lane-16, halves 0-7 -> K=k0+8..15, halves 8-15 -> K=k0+24..31
__device__ __forceinline__ v16h load_frag_rm(const _Float16* __restrict__ base,
                                             long ld, int row0, int k0) {
    int lane = threadIdx.x & 31;
    int r    = row0 + (lane & 15);
    int hs   = (lane >> 4) << 3;            // 0 or 8
    const _Float16* p = base + (long)r * ld + k0 + hs;
    Frag16 f;
    f.p.lo = *(const v8h*)(p);              // K = k0+hs .. +7
    f.p.hi = *(const v8h*)(p + 16);         // K = k0+16+hs .. +7
    return f.v;
}

// ---------------------------------------------------------------------------
// Elementwise f32 -> f16, 8 elements per thread (all sizes are multiples of 8)
// ---------------------------------------------------------------------------
__global__ void cvt_f32_to_f16(const float* __restrict__ in,
                               _Float16* __restrict__ out, long n8) {
    long i = (long)blockIdx.x * blockDim.x + threadIdx.x;
    if (i >= n8) return;
    v4f a = *(const v4f*)(in + i * 8);
    v4f b = *(const v4f*)(in + i * 8 + 4);
    v8h o;
    for (int j = 0; j < 4; ++j) { o[j] = (_Float16)a[j]; o[4 + j] = (_Float16)b[j]; }
    *(v8h*)(out + i * 8) = o;
}

// ---------------------------------------------------------------------------
// vT[bh][d][t] = qkv[(b*T + t)*3D + 2D + h*hd + d], LDS-tiled 64x64 transpose.
// Grid: (B*H) * (T/64), block 256.
// ---------------------------------------------------------------------------
__global__ void transpose_v_tiled(const _Float16* __restrict__ qkv,
                                  _Float16* __restrict__ vT,
                                  int T, int H, int hd) {
    const int D = H * hd;
    const long ld3 = (long)3 * D;
    int nT = T / 64;
    int bh = blockIdx.x / nT;
    int t0 = (blockIdx.x % nT) * 64;
    int b = bh / H, h = bh % H;

    __shared__ _Float16 tile[64][68];       // 68 halves = 34 dwords, conflict-light

    const _Float16* src = qkv + ((long)b * T + t0) * ld3 + 2 * D + h * hd;
    for (int i = 0; i < 16; ++i) {
        int idx = threadIdx.x + i * 256;
        int tl = idx >> 6, dl = idx & 63;   // dl fastest -> coalesced reads
        tile[dl][tl] = src[(long)tl * ld3 + dl];
    }
    __syncthreads();
    _Float16* dst = vT + (long)bh * hd * T + t0;
    for (int i = 0; i < 16; ++i) {
        int idx = threadIdx.x + i * 256;
        int dl = idx >> 6, tl = idx & 63;   // tl fastest -> coalesced writes
        dst[(long)dl * T + tl] = tile[dl][tl];
    }
}

// ---------------------------------------------------------------------------
// C[M,N] = A[M,K] (row-major f16) x B[N,K]^T (row-major f16, torch-Linear W)
// One wave -> 32x32 C tile; 2x2 fragments, 4 WMMAs per K=32 step.
// ---------------------------------------------------------------------------
template <bool OUT_HALF>
__global__ void gemm_wmma(const _Float16* __restrict__ A,
                          const _Float16* __restrict__ B,
                          void* __restrict__ C,
                          int M, int N, int K, int ldc) {
    int wave    = threadIdx.x >> 5;
    int lane    = threadIdx.x & 31;
    int nTilesN = N >> 5;
    int nTiles  = (M >> 5) * nTilesN;
    int tile    = blockIdx.x * (blockDim.x >> 5) + wave;
    if (tile >= nTiles) return;                       // wave-uniform
    int tm = (tile / nTilesN) << 5;
    int tn = (tile % nTilesN) << 5;

    v8f acc[2][2] = {{{0}, {0}}, {{0}, {0}}};

    for (int k = 0; k < K; k += 32) {
        if (k + 32 < K) {                             // uniform branch
            __builtin_prefetch(A + (long)(tm + (lane & 15)) * K + k + 32, 0, 3);
            __builtin_prefetch(B + (long)(tn + (lane & 15)) * K + k + 32, 0, 3);
        }
        v16h a0 = load_frag_rm(A, K, tm,      k);
        v16h a1 = load_frag_rm(A, K, tm + 16, k);
        v16h b0 = load_frag_rm(B, K, tn,      k);
        v16h b1 = load_frag_rm(B, K, tn + 16, k);
        acc[0][0] = wmma_f16(a0, b0, acc[0][0]);
        acc[0][1] = wmma_f16(a0, b1, acc[0][1]);
        acc[1][0] = wmma_f16(a1, b0, acc[1][0]);
        acc[1][1] = wmma_f16(a1, b1, acc[1][1]);
    }

    // C fragment layout: VGPR r holds (row=r, lanes0-15 cols) / (row=r+8, lanes16-31)
    int colb = lane & 15;
    int rowb = (lane >> 4) << 3;
    for (int i = 0; i < 2; ++i)
        for (int j = 0; j < 2; ++j)
            for (int r = 0; r < 8; ++r) {
                long row = tm + i * 16 + rowb + r;
                long col = tn + j * 16 + colb;
                float v  = acc[i][j][r];
                if (OUT_HALF) ((_Float16*)C)[row * ldc + col] = (_Float16)v;
                else          ((float*)C)[row * ldc + col]    = v;
            }
}

// ---------------------------------------------------------------------------
// Flash attention, causal + ALiBi, fully transposed dataflow.
//   qkv : [B*T, 3*D] f16 ;  vT : [B*H, hd, T] f16 ;  outh : [B*T, D] f16
// One wave owns 16 query columns (lane = query); key tiles of 32.
//   S^T = K(16kt x 64) · Q^T(64 x 16q)   -> C-frag: lane=q, reg r = kt r/r+8
//   O^T = V^T(64d x 32kt) · P^T(32 x 16q)-> C-frag: lane=q, reg r = d
// P^T C-frag layout == B-operand layout, so P is repacked in registers.
// All K fragments are issued before the score WMMAs (partial loadcnt waits),
// and V fragments are issued before the softmax math to hide their latency.
// ---------------------------------------------------------------------------
__global__ void attn_flash_wmma(const _Float16* __restrict__ qkv,
                                const _Float16* __restrict__ vT,
                                _Float16* __restrict__ outh,
                                int T, int H, int hd) {
    const int  D  = H * hd;            // 1024
    const long ld = (long)3 * D;       // 3072
    int nQblk = T / 64;
    int bh    = blockIdx.x / nQblk;
    int qblk  = blockIdx.x % nQblk;
    int b = bh / H, h = bh % H;
    int wave = threadIdx.x >> 5;
    int lane = threadIdx.x & 31;
    int qbase = qblk * 64 + wave * 16;

    const float scale = 0.125f;                                   // hd^-0.5
    const float slope = exp2f(-(float)(h + 1) / (float)H);        // ALiBi
    const float NEGINF = -__builtin_inff();

    const _Float16* Qp = qkv + (long)b * T * ld + (long)h * hd;
    const _Float16* Kp = Qp + D;
    const _Float16* Vt = vT + (long)bh * hd * T;

    // Q^T B-operand fragments for this wave's 16 queries (hd=64 -> 2 chunks)
    v16h qfA = load_frag_rm(Qp, ld, qbase, 0);    // dd = 0..31
    v16h qfB = load_frag_rm(Qp, ld, qbase, 32);   // dd = 32..63

    int qlane = qbase + (lane & 15);              // this lane's query index
    int rh8   = (lane >> 4) << 3;                 // 0 or 8 (kt / d row offset)

    // rel = qlane - kt_index, affine per tile: precompute per-register bases
    float relA[8], srelA[8];
    for (int r = 0; r < 8; ++r) {
        float rl  = (float)(qlane - rh8 - r);     // rel at kt = 0, first half
        relA[r]   = rl;
        srelA[r]  = slope * rl;
    }
    const float slope16 = 16.0f * slope;

    v8f  o[4] = {{0}, {0}, {0}, {0}};             // O^T: 64d x 16q, 4 frags
    float mrow = NEGINF, lrow = 0.f;
    float ktf = 0.0f;                             // (float)kt, updated per tile

    for (int kt = 0; kt <= qbase + 15; kt += 32, ktf += 32.0f) {
        // ---- issue all K fragments, then the 4 score WMMAs ----
        v16h kf0a = load_frag_rm(Kp, ld, kt,      0);
        v16h kf0b = load_frag_rm(Kp, ld, kt,      32);
        v16h kf1a = load_frag_rm(Kp, ld, kt + 16, 0);
        v16h kf1b = load_frag_rm(Kp, ld, kt + 16, 32);
        v8f s0 = {0}, s1 = {0};
        s0 = wmma_f16(kf0a, qfA, s0);
        s0 = wmma_f16(kf0b, qfB, s0);
        s1 = wmma_f16(kf1a, qfA, s1);
        s1 = wmma_f16(kf1b, qfB, s1);

        // ---- issue V fragments early; softmax math below hides their latency
        v16h vf0 = load_frag_rm(Vt, T,  0, kt);
        v16h vf1 = load_frag_rm(Vt, T, 16, kt);
        v16h vf2 = load_frag_rm(Vt, T, 32, kt);
        v16h vf3 = load_frag_rm(Vt, T, 48, kt);

        // ---- bias + causal mask + local max (all-float, per-lane regs) ----
        float sk = slope * ktf;                   // slope * kt
        float a0[8], a1[8];
        float mloc = NEGINF;
        for (int r = 0; r < 8; ++r) {
            float rel0  = relA[r] - ktf;          // qlane - k0 (exact integer)
            float bias0 = sk - srelA[r];          // -slope*(qlane-k0)
            float v0 = fmaf(s0[r], scale, bias0);
            float v1 = fmaf(s1[r], scale, bias0 + slope16);
            a0[r] = (rel0 >= 0.0f)  ? v0 : NEGINF;
            a1[r] = (rel0 >= 16.0f) ? v1 : NEGINF;
            mloc  = fmaxf(mloc, fmaxf(a0[r], a1[r]));
        }
        mloc = fmaxf(mloc, __shfl_xor(mloc, 16, 32));   // join the two kt halves
        float mnew = fmaxf(mrow, mloc);

        // ---- exponentials, P^T repacked directly into a B-operand frag ----
        float corr = __expf(mrow - mnew);               // 0 when mrow == -inf
        v16h pb = {};
        float ps = 0.f;
        for (int r = 0; r < 8; ++r) {
            float p0 = __expf(a0[r] - mnew);            // exp(-inf) == 0
            float p1 = __expf(a1[r] - mnew);
            ps += p0 + p1;
            pb[r]     = (_Float16)p0;                   // kk = rh8 + r
            pb[8 + r] = (_Float16)p1;                   // kk = 16 + rh8 + r
        }
        ps += __shfl_xor(ps, 16, 32);
        lrow = lrow * corr + ps;
        mrow = mnew;

        // ---- O^T = corr*O^T + V^T x P^T ----
        o[0] = wmma_f16(vf0, pb, o[0] * corr);
        o[1] = wmma_f16(vf1, pb, o[1] * corr);
        o[2] = wmma_f16(vf2, pb, o[2] * corr);
        o[3] = wmma_f16(vf3, pb, o[3] * corr);
    }

    // ---- normalize, write O^T: lane q stores 8 contiguous d per fragment ----
    float rinv = 1.0f / lrow;
    _Float16* dst = outh + ((long)b * T + qlane) * D + (long)h * hd + rh8;
    for (int f = 0; f < 4; ++f) {
        v8h ov;
        for (int r = 0; r < 8; ++r) ov[r] = (_Float16)(o[f][r] * rinv);
        *(v8h*)(dst + f * 16) = ov;                     // d = f*16 + rh8 + r
    }
}

// ---------------------------------------------------------------------------
// Host launcher
// ---------------------------------------------------------------------------
extern "C" void kernel_launch(void* const* d_in, const int* in_sizes, int n_in,
                              void* d_out, int out_size, void* d_ws, size_t ws_size,
                              hipStream_t stream) {
    (void)in_sizes; (void)n_in; (void)out_size; (void)ws_size;
    constexpr int  Bb = 2, T = 2048, D = 1024, H = 16, HD = 64;
    constexpr long M  = (long)Bb * T;          // 4096

    const float* x     = (const float*)d_in[0];
    const float* w_qkv = (const float*)d_in[1];
    const float* w_out = (const float*)d_in[2];
    float*       out   = (float*)d_out;

    // workspace carve-up (256B aligned)
    char* ws = (char*)d_ws;
    size_t off = 0;
    auto carve = [&](size_t bytes) {
        size_t r = off;
        off += (bytes + 255) & ~(size_t)255;
        return r;
    };
    _Float16* xh    = (_Float16*)(ws + carve(M * D * 2));               //  8 MB
    _Float16* wqkvh = (_Float16*)(ws + carve((size_t)3 * D * D * 2));   //  6 MB
    _Float16* wouth = (_Float16*)(ws + carve((size_t)D * D * 2));       //  2 MB
    _Float16* qkvh  = (_Float16*)(ws + carve(M * 3 * D * 2));           // 24 MB
    _Float16* vTh   = (_Float16*)(ws + carve((size_t)Bb * H * HD * T * 2)); // 8 MB
    _Float16* atth  = (_Float16*)(ws + carve(M * D * 2));               //  8 MB

    // 1) f32 -> f16 conversions (8 elements / thread)
    {
        long n1 = M * D / 8, n2 = (long)3 * D * D / 8, n3 = (long)D * D / 8;
        cvt_f32_to_f16<<<dim3((n1 + 255) / 256), dim3(256), 0, stream>>>(x, xh, n1);
        cvt_f32_to_f16<<<dim3((n2 + 255) / 256), dim3(256), 0, stream>>>(w_qkv, wqkvh, n2);
        cvt_f32_to_f16<<<dim3((n3 + 255) / 256), dim3(256), 0, stream>>>(w_out, wouth, n3);
    }

    // 2) qkv = x @ w_qkv^T   (4096x1024 x 1024x3072) -> f16 [4096,3072]
    {
        int tiles = (int)(M / 32) * (3 * D / 32);     // 12288
        gemm_wmma<true><<<dim3(tiles / 8), dim3(256), 0, stream>>>(
            xh, wqkvh, qkvh, (int)M, 3 * D, D, 3 * D);
    }

    // 3) transpose V per head (LDS-tiled): vT[bh][d][t]
    {
        int blocks = Bb * H * (T / 64);               // 1024
        transpose_v_tiled<<<dim3(blocks), dim3(256), 0, stream>>>(qkvh, vTh, T, H, HD);
    }

    // 4) flash attention (causal + ALiBi, transposed dataflow) -> atth f16
    {
        int blocks = Bb * H * (T / 64);               // 1024
        attn_flash_wmma<<<dim3(blocks), dim3(128), 0, stream>>>(qkvh, vTh, atth, T, H, HD);
    }

    // 5) out = attn @ w_out^T -> f32 d_out
    {
        int tiles = (int)(M / 32) * (D / 32);         // 4096
        gemm_wmma<false><<<dim3(tiles / 8), dim3(256), 0, stream>>>(
            atth, wouth, out, (int)M, D, D, D);
    }
}
